// DenseCoordination_84593675862307
// MI455X (gfx1250) — matrix operand order repaired and checked by reading, not compiled, not run on hardware
//
#include <hip/hip_runtime.h>

typedef __attribute__((ext_vector_type(16))) __bf16 v16bf;
typedef __attribute__((ext_vector_type(8)))  float  v8f;

union Frag {
    v16bf v;
    uint4 q[2];
};

__device__ __forceinline__ unsigned short f32_to_bf16(float f) {
    unsigned u = __float_as_uint(f);
    u += 0x7fffu + ((u >> 16) & 1u);      // round-to-nearest-even
    return (unsigned short)(u >> 16);
}

#define Bn 4
#define Nn 256
#define Dn 256
#define Hn 512
#define LSTR 264   // padded LDS row stride in halves (528 B: 16B aligned, bank-spread)

// ---------------------------------------------------------------------------
// Prep 1: hi_r[bi,h] = s[bi,:]@W1a[:,h] + b1[h];  hj[bi,h] = s[bi,:]@W1b[:,h]
// ---------------------------------------------------------------------------
__global__ __launch_bounds__(256)
void prep_hihj(const float* __restrict__ s, const float* __restrict__ W1,
               const float* __restrict__ b1,
               float* __restrict__ hi_r, float* __restrict__ hj) {
    __shared__ float srow[Dn];
    const int blk = blockIdx.x;            // b*N + i
    const int tid = threadIdx.x;
    srow[tid] = s[(size_t)blk * Dn + tid];
    __syncthreads();
    const int h0 = tid, h1 = tid + 256;
    float a0 = 0.f, a1 = 0.f, c0 = 0.f, c1 = 0.f;
    for (int d = 0; d < Dn; ++d) {
        const float sv = srow[d];
        const float* ra = W1 + (size_t)d * Hn;            // W1a row d
        const float* rb = W1 + (size_t)(Dn + d) * Hn;     // W1b row d
        a0 += sv * ra[h0];
        a1 += sv * ra[h1];
        c0 += sv * rb[h0];
        c1 += sv * rb[h1];
    }
    const size_t o = (size_t)blk * Hn;
    hi_r[o + h0] = a0 + b1[h0];
    hi_r[o + h1] = a1 + b1[h1];
    hj[o + h0]   = c0;
    hj[o + h1]   = c1;
}

// ---------------------------------------------------------------------------
// Prep 2: W1cT[h][d] = bf16(W1[2D+d][h])   (512 x 256 bf16)
// ---------------------------------------------------------------------------
__global__ __launch_bounds__(256)
void prep_w1ct(const float* __restrict__ W1, unsigned short* __restrict__ w1ct) {
    const int d = blockIdx.x;
    const int tid = threadIdx.x;
#pragma unroll
    for (int hh = 0; hh < 2; ++hh) {
        const int h = tid + hh * 256;
        const float v = W1[(size_t)(2 * Dn + d) * Hn + h];
        w1ct[(size_t)h * Dn + d] = f32_to_bf16(v);
    }
}

// ---------------------------------------------------------------------------
// Main: one workgroup per (b,i).  X = s_j*s_i (bf16, LDS); B h-tiles staged
// into LDS via async-to-LDS (double buffered); hij = X@W1c via
// v_wmma_f32_16x16x32_bf16; fused relu+W2 dot -> logits; masked softmax;
// ctx = w@s; gate = 1.
// ---------------------------------------------------------------------------
__global__ __launch_bounds__(256)
void pair_kernel(const float* __restrict__ s,
                 const float* __restrict__ hi_r, const float* __restrict__ hj,
                 const unsigned short* __restrict__ w1ct,
                 const float* __restrict__ W2, const float* __restrict__ b2,
                 const int* __restrict__ adj, const int* __restrict__ active_m,
                 const int* __restrict__ act_m,
                 float* __restrict__ ctx_out, float* __restrict__ gate_out,
                 float* __restrict__ w_out) {
    __shared__ unsigned short Xs[Nn * LSTR];        // 135168 B
    __shared__ unsigned short Bs[2][16 * LSTR];     // 2 x 8448 B (double buffer)
    __shared__ float Lg[Nn];
    __shared__ float Wrow[Nn];
    __shared__ float red[8];

    const int blk  = blockIdx.x;
    const int b    = blk >> 8;
    const int i    = blk & 255;
    const int tid  = threadIdx.x;
    const int lane = tid & 31;
    const int wave = tid >> 5;

    const float* sb = s + (size_t)b * Nn * Dn;

    // ---- Phase 0: X[j,d] = s[b,j,d]*s[b,i,d] in bf16, coalesced over d=tid
    const float si_d = sb[(size_t)i * Dn + tid];
    for (int j = 0; j < Nn; ++j) {
        Xs[j * LSTR + tid] = f32_to_bf16(sb[(size_t)j * Dn + tid] * si_d);
    }

    // async stage of B h-tile `ht` into Bs[buf]: 16 rows x 512B, 512 x 16B chunks
    auto stage = [&](int ht, int buf) {
#pragma unroll
        for (int r2 = 0; r2 < 2; ++r2) {
            const int k   = tid + r2 * 256;
            const int row = k >> 5;
            const int cc  = (k & 31) * 8;            // half offset within row
            const unsigned short* gp = w1ct + (size_t)(ht * 16 + row) * Dn + cc;
            const unsigned loff =
                (unsigned)(size_t)(const void*)&Bs[buf][row * LSTR + cc];
            const unsigned long long ga = (unsigned long long)(size_t)gp;
            asm volatile("global_load_async_to_lds_b128 %0, %1, off"
                         :: "v"(loff), "v"(ga) : "memory");
        }
    };

    stage(0, 0);
    asm volatile("s_wait_asynccnt 0x0" ::: "memory");
    __syncthreads();   // Xs complete + Bs[0] staged

    // ---- Phase 1: tiled WMMA GEMM + fused relu/W2 epilogue
    const int nl    = lane & 15;
    const int koffA = (lane < 16) ? 0 : 8;    // A: K chunks {koffA..+8, 16+koffA..+8}
    const int koffB = (lane < 16) ? 0 : 8;    // B: K chunks {koffB..+8, 16+koffB..+8}
    const int moff  = (lane < 16) ? 0 : 8;    // C: row offset for this half-wave

    const float* hjb = hj + (size_t)(b * Nn) * Hn;
    const size_t ro  = (size_t)(b * Nn + i) * Hn;

    // cache A fragments for both owned j-tiles (K = 0..255)
    Frag af[2][8];
#pragma unroll
    for (int jt2 = 0; jt2 < 2; ++jt2) {
        const int rowA = (wave + jt2 * 8) * 16 + nl;
        const unsigned short* abase = &Xs[rowA * LSTR + koffA];
#pragma unroll
        for (int kb = 0; kb < 8; ++kb) {
            af[jt2][kb].q[0] = *(const uint4*)(abase + kb * 32);
            af[jt2][kb].q[1] = *(const uint4*)(abase + kb * 32 + 16);
        }
    }

    float plog[2][8];
#pragma unroll
    for (int jt2 = 0; jt2 < 2; ++jt2)
#pragma unroll
        for (int v = 0; v < 8; ++v) plog[jt2][v] = 0.f;

    for (int ht = 0; ht < 32; ++ht) {
        if (ht + 1 < 32) stage(ht + 1, (ht + 1) & 1);   // overlap with WMMAs

        const unsigned short* bb = &Bs[ht & 1][nl * LSTR + koffB];
        const int   hh  = ht * 16 + nl;
        const float rv  = hi_r[ro + hh];
        const float w2v = W2[hh];

#pragma unroll
        for (int jt2 = 0; jt2 < 2; ++jt2) {
            v8f c = {0.f, 0.f, 0.f, 0.f, 0.f, 0.f, 0.f, 0.f};
#pragma unroll
            for (int kb = 0; kb < 8; ++kb) {
                Frag bf;
                bf.q[0] = *(const uint4*)(bb + kb * 32);
                bf.q[1] = *(const uint4*)(bb + kb * 32 + 16);
                c = __builtin_amdgcn_wmma_f32_16x16x32_bf16(
                        false, af[jt2][kb].v, false, bf.v, (short)0, c,
                        false, false);
            }
            // epilogue: h = relu(c + hi_i + hj_j + b1),  plog += h * W2[h]
            const int jb = (wave + jt2 * 8) * 16 + moff;
#pragma unroll
            for (int v = 0; v < 8; ++v) {
                float t = c[v] + rv + hjb[(size_t)(jb + v) * Hn + hh];
                t = t > 0.f ? t : 0.f;
                plog[jt2][v] += t * w2v;
            }
        }

        asm volatile("s_wait_asynccnt 0x0" ::: "memory");
        __syncthreads();   // Bs[(ht+1)&1] ready; Bs[ht&1] free to overwrite
    }

    // reduce over the 16 lanes of each half-wave (h classes) -> logits
#pragma unroll
    for (int jt2 = 0; jt2 < 2; ++jt2) {
        const int jt = wave + jt2 * 8;
#pragma unroll
        for (int v = 0; v < 8; ++v) {
            float p = plog[jt2][v];
            p += __shfl_xor(p, 8, 32);
            p += __shfl_xor(p, 4, 32);
            p += __shfl_xor(p, 2, 32);
            p += __shfl_xor(p, 1, 32);
            if (nl == 0) Lg[jt * 16 + moff + v] = p;
        }
    }
    __syncthreads();

    // ---- Phase 2: masked softmax over j (row fully inside this workgroup)
    const int j = tid;
    const bool valid = (act_m[b * Nn + i] > 0) &&
                       (adj[(size_t)(b * Nn + i) * Nn + j] > 0) &&
                       (active_m[b * Nn + j] > 0);
    const float NEGINF = -__builtin_inff();
    const float lg = valid ? (Lg[j] + b2[0]) : NEGINF;

    float m = lg;
#pragma unroll
    for (int o = 16; o >= 1; o >>= 1) m = fmaxf(m, __shfl_xor(m, o, 32));
    if (lane == 0) red[wave] = m;
    __syncthreads();
    float mall = red[0];
#pragma unroll
    for (int k = 1; k < 8; ++k) mall = fmaxf(mall, red[k]);

    const float e = (lg > NEGINF) ? __expf(lg - mall) : 0.f;
    float ssum = e;
#pragma unroll
    for (int o = 16; o >= 1; o >>= 1) ssum += __shfl_xor(ssum, o, 32);
    __syncthreads();                 // done reading red[] for max
    if (lane == 0) red[wave] = ssum;
    __syncthreads();
    float tot = red[0];
#pragma unroll
    for (int k = 1; k < 8; ++k) tot += red[k];

    const float wv = (tot > 0.f) ? (e / tot) : 0.f;   // nan_to_num path
    const size_t orow = (size_t)(b * Nn + i) * Nn;
    w_out[orow + j]    = wv;
    gate_out[orow + j] = 1.0f;
    Wrow[j] = wv;
    __syncthreads();

    // ---- Phase 3: ctx[b,i,d] = sum_j w_j * s[b,j,d]  (coalesced over d=tid)
    float acc = 0.f;
    const int d = tid;
    for (int jj = 0; jj < Nn; ++jj) acc += Wrow[jj] * sb[(size_t)jj * Dn + d];
    ctx_out[(size_t)(b * Nn + i) * Dn + d] = acc;
}

// ---------------------------------------------------------------------------
extern "C" void kernel_launch(void* const* d_in, const int* in_sizes, int n_in,
                              void* d_out, int out_size, void* d_ws, size_t ws_size,
                              hipStream_t stream) {
    const float* s  = (const float*)d_in[0];
    const float* W1 = (const float*)d_in[1];
    const float* b1 = (const float*)d_in[2];
    const float* W2 = (const float*)d_in[3];
    const float* b2 = (const float*)d_in[4];
    const int* adj      = (const int*)d_in[5];
    const int* active_m = (const int*)d_in[6];
    const int* act_m    = (const int*)d_in[7];

    float* ctx_out  = (float*)d_out;                 // B*N*D
    float* gate_out = ctx_out + Bn * Nn * Nn;        // B*N*N
    float* w_out    = gate_out + Bn * Nn * Nn;       // B*N*N

    char* ws = (char*)d_ws;
    float* hi_r = (float*)ws;                                    // 2 MB
    float* hjw  = (float*)(ws + (size_t)Bn * Nn * Hn * 4);       // 2 MB
    unsigned short* w1ct =
        (unsigned short*)(ws + (size_t)2 * Bn * Nn * Hn * 4);    // 256 KB

    prep_hihj<<<Bn * Nn, 256, 0, stream>>>(s, W1, b1, hi_r, hjw);
    prep_w1ct<<<Dn, 256, 0, stream>>>(W1, w1ct);
    pair_kernel<<<Bn * Nn, 256, 0, stream>>>(s, hi_r, hjw, w1ct, W2, b2,
                                             adj, active_m, act_m,
                                             ctx_out, gate_out, w_out);
}
